// DecoderLayer_59296318488701
// MI455X (gfx1250) — matrix operand order repaired and checked by reading, not compiled, not run on hardware
//
#include <hip/hip_runtime.h>
#include <hip/hip_bf16.h>
#include <math.h>
#include <stdint.h>

typedef __attribute__((ext_vector_type(16))) _Float16 v16h;
typedef __attribute__((ext_vector_type(8)))  _Float16 v8h;
typedef __attribute__((ext_vector_type(8)))  float    v8f;
typedef __attribute__((ext_vector_type(4)))  unsigned int v4u;
typedef __attribute__((ext_vector_type(8)))  int v8i;
typedef __attribute__((ext_vector_type(4)))  int v4i;

#define S_LEN 2048
#define H_DIM 2048
#define NH_   16
#define DQK_  192
#define DV_   128
#define QR_   768
#define KVR_  512
#define NE_   16
#define MMOE_ 1024

__device__ __forceinline__ v16h cat8(v8h lo, v8h hi) {
  return __builtin_shufflevector(lo, hi, 0,1,2,3,4,5,6,7,8,9,10,11,12,13,14,15);
}
__device__ __forceinline__ v8f zero8() {
  v8f z = {0.f,0.f,0.f,0.f,0.f,0.f,0.f,0.f};
  return z;
}

// --- CDNA5 async copy global->LDS (ASYNCcnt), per-lane LDS dest ------------
__device__ __forceinline__ void async_load_b128(const void* gbase, int byte_off,
                                                void* lds_ptr) {
  uint32_t lds = (uint32_t)(uintptr_t)lds_ptr;   // low 32 bits = LDS offset
  asm volatile("global_load_async_to_lds_b128 %0, %1, %2"
               :: "v"(lds), "v"(byte_off),
                  "s"((unsigned long long)(uintptr_t)gbase)
               : "memory");
}
__device__ __forceinline__ void wait_async0() {
  asm volatile("s_wait_asynccnt 0x0" ::: "memory");
}

// --- CDNA5 Tensor Data Mover: 2D tile (128 rows x 32 halves) -> padded LDS -
// D# per ISA 08_async_tensor §8.3/8.4:
//   group0: count=1 | lds_addr | global_addr(57b) | type=2
//   group1: data_size=2B, pad_enable, pad_interval=3 (16 DW), pad_amount=3
//           (4 DW) -> 80B LDS row pitch (== LDST halves); tensor_dim0=32,
//           tensor_dim1=128, tile_dim0=32, tile_dim1=128, dim0_stride=K.
__device__ __forceinline__ void tdm_load_tile_128x32(const _Float16* gsrc,
                                                     void* lds_dst, int K) {
  unsigned long long ga = (unsigned long long)(uintptr_t)gsrc;
  v4u g0;
  g0[0] = 1u;                                          // count=1 (valid, user)
  g0[1] = (uint32_t)(uintptr_t)lds_dst;                // lds_addr (bytes)
  g0[2] = (uint32_t)(ga & 0xFFFFFFFFu);                // global_addr[31:0]
  g0[3] = (uint32_t)((ga >> 32) & 0x1FFFFFFu) | (2u << 30);  // [56:32]|type=2
  v8i g1;
  g1[0] = (int)((1u << 16) | (1u << 20) | (3u << 22) | (3u << 25));
  g1[1] = (int)(32u << 16);        // tensor_dim0 = 32 (low 16)
  g1[2] = (int)(128u << 16);       // tensor_dim0 hi=0 | tensor_dim1 = 128
  g1[3] = (int)(32u << 16);        // tensor_dim1 hi=0 | tile_dim0 = 32
  g1[4] = 128;                     // tile_dim1 = 128, tile_dim2 = 0
  g1[5] = K;                       // tensor_dim0_stride (elements), low 32
  g1[6] = 0;
  g1[7] = 0;
  v4i g2 = {0,0,0,0};
  v4i g3 = {0,0,0,0};
#if defined(__clang_major__) && __clang_major__ >= 23
  v8i g4 = {0,0,0,0,0,0,0,0};
  __builtin_amdgcn_tensor_load_to_lds(g0, g1, g2, g3, g4, 0);
#else
  __builtin_amdgcn_tensor_load_to_lds(g0, g1, g2, g3, 0);
#endif
}

// ---------------------------------------------------------------------------
// NT GEMM, f16 operands, fp32 accumulate: C[M,N] = A[M,K] * B[N,K]^T
// A tiles: per-lane GLOBAL_LOAD_ASYNC_TO_LDS_B128 (ASYNCcnt).
// B tiles: one TENSOR_LOAD_TO_LDS per k-step issued by wave 0 (TENSORcnt).
// Both double-buffered against the WMMA stream.
// mode 0: C = AB^T ; mode 1: C = res + AB^T ; mode 2: C += AB^T
// ---------------------------------------------------------------------------
#define GBM 128
#define GBN 128
#define GBK 32
#define LDST 40   // LDS row stride in halves (32 + 8 pad, == TDM pad layout)

__global__ __launch_bounds__(256) void gemm_nt_f16_kernel(
    const _Float16* __restrict__ A, const _Float16* __restrict__ B,
    float* __restrict__ C, const float* __restrict__ res,
    int M, int N, int K, int mode)
{
  __shared__ _Float16 Ah[2][GBM * LDST];
  __shared__ _Float16 Bh[2][GBN * LDST];
  const int tid  = threadIdx.x;
  const int wid  = tid >> 5;
  const int lane = tid & 31;
  const int lhs  = lane >> 4;
  const int ll   = lane & 15;
  const int bm = blockIdx.y * GBM;
  const int bn = blockIdx.x * GBN;
  const int wm = (wid & 3) * 32;
  const int wn = (wid >> 2) * 64;

  v8f acc[2][4];
  #pragma unroll
  for (int i = 0; i < 2; ++i)
    #pragma unroll
    for (int j = 0; j < 4; ++j) acc[i][j] = zero8();

  auto issue = [&](int kb, int bsel) {
    #pragma unroll
    for (int j = 0; j < 2; ++j) {          // A: 2 x 16B per thread
      int id = tid + j * 256;
      int row = id >> 2, seg = (id & 3) * 8;
      async_load_b128(A, (int)((((size_t)(bm + row)) * K + kb + seg) * 2),
                      &Ah[bsel][row * LDST + seg]);
    }
    if (wid == 0)                           // B: one TDM descriptor
      tdm_load_tile_128x32(B + (size_t)bn * K + kb, &Bh[bsel][0], K);
  };

  issue(0, 0);
  wait_async0();
  __builtin_amdgcn_s_wait_tensorcnt(0);
  __syncthreads();

  int cur = 0;
  for (int kb = 0; kb < K; kb += GBK) {
    if (kb + GBK < K) issue(kb + GBK, cur ^ 1);   // overlap copies with math

    const _Float16* Ac = &Ah[cur][0];
    const _Float16* Bc = &Bh[cur][0];
    v16h af[2], bf[4];
    #pragma unroll
    for (int ms = 0; ms < 2; ++ms) {
      int r = wm + ms * 16 + ll;
      const v8h lo = *(const v8h*)&Ac[r * LDST + lhs * 8];
      const v8h hi = *(const v8h*)&Ac[r * LDST + 16 + lhs * 8];
      af[ms] = cat8(lo, hi);
    }
    #pragma unroll
    for (int ns = 0; ns < 4; ++ns) {
      int c = wn + ns * 16 + ll;
      const v8h lo = *(const v8h*)&Bc[c * LDST + lhs * 8];
      const v8h hi = *(const v8h*)&Bc[c * LDST + 16 + lhs * 8];
      bf[ns] = cat8(lo, hi);
    }
    #pragma unroll
    for (int ms = 0; ms < 2; ++ms)
      #pragma unroll
      for (int ns = 0; ns < 4; ++ns)
        acc[ms][ns] = __builtin_amdgcn_wmma_f32_16x16x32_f16(
            false, af[ms], false, bf[ns], (short)0, acc[ms][ns], false, false);

    wait_async0();                       // next A tile landed
    __builtin_amdgcn_s_wait_tensorcnt(0);// next B tile landed (wave 0)
    __syncthreads();
    cur ^= 1;
  }

  #pragma unroll
  for (int ms = 0; ms < 2; ++ms)
    #pragma unroll
    for (int ns = 0; ns < 4; ++ns)
      #pragma unroll
      for (int r = 0; r < 8; ++r) {
        int m = bm + wm + ms * 16 + lhs * 8 + r;
        int n = bn + wn + ns * 16 + ll;
        size_t idx = (size_t)m * N + n;
        float v = acc[ms][ns][r];
        if (mode == 1) v += res[idx];
        else if (mode == 2) v += C[idx];
        C[idx] = v;
      }
}

// ---------------------------------------------------------------------------
// Flash attention for MLA head layout (fp32 in, f16 ctx out).
// ---------------------------------------------------------------------------
#define KST 200
#define VST 72
#define PST 72

__global__ __launch_bounds__(256) void flash_mla_kernel(
    const float* __restrict__ Q, const float* __restrict__ KV,
    _Float16* __restrict__ O16, float scale)
{
  __shared__ _Float16 Kb[64 * KST];
  __shared__ _Float16 Vt[128 * VST];
  __shared__ _Float16 Pl[8 * 16 * PST];
  const int h  = blockIdx.x;
  const int qb = blockIdx.y;
  const int tid = threadIdx.x, wid = tid >> 5, lane = tid & 31;
  const int lhs = lane >> 4, ll = lane & 15;

  v16h qf[6];
  {
    const int qr = qb * 128 + wid * 16 + ll;
    const float* qp = Q + (size_t)qr * (NH_ * DQK_) + h * DQK_;
    #pragma unroll
    for (int f = 0; f < 6; ++f) {
      const float4 a0 = *(const float4*)(qp + f * 32 + lhs * 8);
      const float4 a1 = *(const float4*)(qp + f * 32 + lhs * 8 + 4);
      const float4 b0 = *(const float4*)(qp + f * 32 + 16 + lhs * 8);
      const float4 b1 = *(const float4*)(qp + f * 32 + 16 + lhs * 8 + 4);
      float t[16];
      t[0]=a0.x; t[1]=a0.y; t[2]=a0.z; t[3]=a0.w;
      t[4]=a1.x; t[5]=a1.y; t[6]=a1.z; t[7]=a1.w;
      t[8]=b0.x; t[9]=b0.y; t[10]=b0.z; t[11]=b0.w;
      t[12]=b1.x; t[13]=b1.y; t[14]=b1.z; t[15]=b1.w;
      v16h q;
      #pragma unroll
      for (int i = 0; i < 16; ++i) q[i] = (_Float16)(t[i] * scale);
      qf[f] = q;
    }
  }

  v8f acc[8];
  float mrow[8], lrow[8];
  #pragma unroll
  for (int d = 0; d < 8; ++d) acc[d] = zero8();
  #pragma unroll
  for (int r = 0; r < 8; ++r) { mrow[r] = -3e38f; lrow[r] = 0.f; }

  const int kbmax = 2 * qb + 1;
  for (int kb = 0; kb <= kbmax; ++kb) {
    #pragma unroll
    for (int jb = 0; jb < 3; ++jb) {
      float4 t[4]; int rows[4], cols[4];
      #pragma unroll
      for (int j = 0; j < 4; ++j) {
        int id = tid + (jb * 4 + j) * 256;
        rows[j] = id / 48; cols[j] = (id % 48) * 4;
        t[j] = *(const float4*)(KV + (size_t)(kb * 64 + rows[j]) * (NH_ * (DQK_ + DV_))
                                + h * (DQK_ + DV_) + cols[j]);
      }
      #pragma unroll
      for (int j = 0; j < 4; ++j) {
        _Float16* d = &Kb[rows[j] * KST + cols[j]];
        d[0] = (_Float16)t[j].x; d[1] = (_Float16)t[j].y;
        d[2] = (_Float16)t[j].z; d[3] = (_Float16)t[j].w;
      }
    }
    #pragma unroll
    for (int jb = 0; jb < 2; ++jb) {
      float4 t[4]; int rows[4], cols[4];
      #pragma unroll
      for (int j = 0; j < 4; ++j) {
        int id = tid + (jb * 4 + j) * 256;
        rows[j] = id >> 5; cols[j] = (id & 31) * 4;
        t[j] = *(const float4*)(KV + (size_t)(kb * 64 + rows[j]) * (NH_ * (DQK_ + DV_))
                                + h * (DQK_ + DV_) + DQK_ + cols[j]);
      }
      #pragma unroll
      for (int j = 0; j < 4; ++j) {
        Vt[(cols[j] + 0) * VST + rows[j]] = (_Float16)t[j].x;
        Vt[(cols[j] + 1) * VST + rows[j]] = (_Float16)t[j].y;
        Vt[(cols[j] + 2) * VST + rows[j]] = (_Float16)t[j].z;
        Vt[(cols[j] + 3) * VST + rows[j]] = (_Float16)t[j].w;
      }
    }
    __syncthreads();

    v8f s[4];
    #pragma unroll
    for (int ns = 0; ns < 4; ++ns) s[ns] = zero8();
    #pragma unroll
    for (int f = 0; f < 6; ++f) {
      #pragma unroll
      for (int ns = 0; ns < 4; ++ns) {
        int c = ns * 16 + ll;
        v8h lo = *(const v8h*)&Kb[c * KST + f * 32 + lhs * 8];
        v8h hi = *(const v8h*)&Kb[c * KST + f * 32 + 16 + lhs * 8];
        s[ns] = __builtin_amdgcn_wmma_f32_16x16x32_f16(
            false, qf[f], false, cat8(lo, hi), (short)0, s[ns], false, false);
      }
    }

    const int qrc = qb * 128 + wid * 16 + lhs * 8;
    #pragma unroll
    for (int ns = 0; ns < 4; ++ns) {
      int jcol = kb * 64 + ns * 16 + ll;
      #pragma unroll
      for (int r = 0; r < 8; ++r)
        if (jcol > qrc + r) s[ns][r] = -1e30f;
    }

    #pragma unroll
    for (int r = 0; r < 8; ++r) {
      float mx = fmaxf(fmaxf(s[0][r], s[1][r]), fmaxf(s[2][r], s[3][r]));
      mx = fmaxf(mx, __shfl_xor(mx, 1, 32));
      mx = fmaxf(mx, __shfl_xor(mx, 2, 32));
      mx = fmaxf(mx, __shfl_xor(mx, 4, 32));
      mx = fmaxf(mx, __shfl_xor(mx, 8, 32));
      float mn = fmaxf(mrow[r], mx);
      float corr = __expf(mrow[r] - mn);
      mrow[r] = mn;
      float sum = 0.f;
      #pragma unroll
      for (int ns = 0; ns < 4; ++ns) {
        float p = __expf(s[ns][r] - mn);
        s[ns][r] = p;
        sum += p;
      }
      sum += __shfl_xor(sum, 1, 32);
      sum += __shfl_xor(sum, 2, 32);
      sum += __shfl_xor(sum, 4, 32);
      sum += __shfl_xor(sum, 8, 32);
      lrow[r] = lrow[r] * corr + sum;
      #pragma unroll
      for (int d = 0; d < 8; ++d) acc[d][r] *= corr;
    }

    #pragma unroll
    for (int ns = 0; ns < 4; ++ns)
      #pragma unroll
      for (int r = 0; r < 8; ++r)
        Pl[wid * 16 * PST + (lhs * 8 + r) * PST + ns * 16 + ll] = (_Float16)s[ns][r];
    __syncthreads();

    #pragma unroll
    for (int ks = 0; ks < 2; ++ks) {
      v8h plo = *(const v8h*)&Pl[wid * 16 * PST + ll * PST + ks * 32 + lhs * 8];
      v8h phi = *(const v8h*)&Pl[wid * 16 * PST + ll * PST + ks * 32 + 16 + lhs * 8];
      v16h pf = cat8(plo, phi);
      #pragma unroll
      for (int d = 0; d < 8; ++d) {
        int c = d * 16 + ll;
        v8h blo = *(const v8h*)&Vt[c * VST + ks * 32 + lhs * 8];
        v8h bhi = *(const v8h*)&Vt[c * VST + ks * 32 + 16 + lhs * 8];
        acc[d] = __builtin_amdgcn_wmma_f32_16x16x32_f16(
            false, pf, false, cat8(blo, bhi), (short)0, acc[d], false, false);
      }
    }
    __syncthreads();
  }

  #pragma unroll
  for (int d = 0; d < 8; ++d)
    #pragma unroll
    for (int r = 0; r < 8; ++r) {
      int m = qb * 128 + wid * 16 + lhs * 8 + r;
      int n = h * DV_ + d * 16 + ll;
      O16[(size_t)m * (NH_ * DV_) + n] = (_Float16)(acc[d][r] / lrow[r]);
    }
}

// ---------------------------------------------------------------------------
__global__ __launch_bounds__(256) void cvt_f16_kernel(
    const float* __restrict__ in, _Float16* __restrict__ out, int n)
{
  int i = (blockIdx.x * 256 + threadIdx.x) * 8;
  if (i >= n) return;
  float4 a = *(const float4*)(in + i);
  float4 b = *(const float4*)(in + i + 4);
  v8h o;
  o[0] = (_Float16)a.x; o[1] = (_Float16)a.y;
  o[2] = (_Float16)a.z; o[3] = (_Float16)a.w;
  o[4] = (_Float16)b.x; o[5] = (_Float16)b.y;
  o[6] = (_Float16)b.z; o[7] = (_Float16)b.w;
  *(v8h*)(out + i) = o;
}

__global__ __launch_bounds__(256) void rmsnorm_f16_kernel(
    const float* __restrict__ x, const float* __restrict__ w,
    _Float16* __restrict__ o, int D)
{
  const int t = blockIdx.x;
  const float* xr = x + (size_t)t * D;
  float ss = 0.f;
  for (int i = threadIdx.x; i < D; i += 256) { float v = xr[i]; ss += v * v; }
  #pragma unroll
  for (int m = 1; m < 32; m <<= 1) ss += __shfl_xor(ss, m, 32);
  __shared__ float red[8];
  if ((threadIdx.x & 31) == 0) red[threadIdx.x >> 5] = ss;
  __syncthreads();
  float tot = 0.f;
  #pragma unroll
  for (int i = 0; i < 8; ++i) tot += red[i];
  float inv = rsqrtf(tot / (float)D + 1e-6f);
  _Float16* orow = o + (size_t)t * D;
  for (int i = threadIdx.x; i < D; i += 256)
    orow[i] = (_Float16)(xr[i] * inv * w[i]);
}

__global__ __launch_bounds__(64) void rope_kernel(
    float* __restrict__ qbuf, float* __restrict__ kvbuf)
{
  const int h = blockIdx.x, s = blockIdx.y;
  const int d = threadIdx.x & 31;
  const bool isK = threadIdx.x >= 32;
  float inv_freq = __powf(10000.0f, -(float)d / 32.0f);
  float sn, cs;
  __sincosf((float)s * inv_freq, &sn, &cs);
  float* p = isK ? (kvbuf + (size_t)s * (NH_ * (DQK_ + DV_)) + h * (DQK_ + DV_) + 128)
                 : (qbuf  + (size_t)s * (NH_ * DQK_) + h * DQK_ + 128);
  float x1 = p[d], x2 = p[d + 32];
  p[d]      = x1 * cs - x2 * sn;
  p[d + 32] = x2 * cs + x1 * sn;
}

__global__ __launch_bounds__(128) void router_kernel(
    const _Float16* __restrict__ h2, const float* __restrict__ rw,
    const float* __restrict__ rb, float* __restrict__ comb, int T, int D)
{
  int t = blockIdx.x * 128 + threadIdx.x;
  if (t >= T) return;
  float acc[NE_];
  #pragma unroll
  for (int e = 0; e < NE_; ++e) acc[e] = rb[e];
  const _Float16* xr = h2 + (size_t)t * D;
  for (int k = 0; k < D; ++k) {
    float xv = (float)xr[k];
    #pragma unroll
    for (int e = 0; e < NE_; ++e) acc[e] += xv * rw[(size_t)e * D + k];
  }
  float prob[NE_];
  #pragma unroll
  for (int e = 0; e < NE_; ++e) prob[e] = 1.f / (1.f + __expf(-acc[e]));
  float w4[4]; int i4[4]; unsigned used = 0;
  #pragma unroll
  for (int s = 0; s < 4; ++s) {
    float best = -1.f; int bi = 0;
    #pragma unroll
    for (int e = 0; e < NE_; ++e)
      if (!((used >> e) & 1) && prob[e] > best) { best = prob[e]; bi = e; }
    w4[s] = best; i4[s] = bi; used |= 1u << bi;
  }
  float sum = w4[0] + w4[1] + w4[2] + w4[3] + 1e-9f;
  float out[NE_];
  #pragma unroll
  for (int e = 0; e < NE_; ++e) out[e] = 0.f;
  #pragma unroll
  for (int s = 0; s < 4; ++s) out[i4[s]] = w4[s] / sum * 2.5f;
  #pragma unroll
  for (int e = 0; e < NE_; ++e) comb[(size_t)t * NE_ + e] = out[e];
}

__global__ __launch_bounds__(256) void silu_mul_kernel(
    const float* __restrict__ g, const float* __restrict__ u,
    const float* __restrict__ comb, int e, _Float16* __restrict__ act,
    int n, int Md)
{
  int i = blockIdx.x * 256 + threadIdx.x;
  if (i >= n) return;
  float gv = g[i], uv = u[i];
  float v = gv / (1.f + __expf(-gv)) * uv;
  if (comb) v *= comb[(size_t)(i / Md) * NE_ + e];
  act[i] = (_Float16)v;
}

// ---------------------------------------------------------------------------
extern "C" void kernel_launch(void* const* d_in, const int* in_sizes, int n_in,
                              void* d_out, int out_size, void* d_ws, size_t ws_size,
                              hipStream_t stream)
{
  (void)in_sizes; (void)n_in; (void)out_size; (void)ws_size;
  const float* x        = (const float*)d_in[0];
  const float* ln1_w    = (const float*)d_in[2];
  const float* ln2_w    = (const float*)d_in[3];
  const float* q_a_w    = (const float*)d_in[4];
  const float* q_a_ln   = (const float*)d_in[5];
  const float* q_b_w    = (const float*)d_in[6];
  const float* kv_a_w   = (const float*)d_in[7];
  const float* kv_a_ln  = (const float*)d_in[8];
  const float* kv_b_w   = (const float*)d_in[9];
  const float* o_w      = (const float*)d_in[10];
  const float* router_w = (const float*)d_in[11];
  const float* router_b = (const float*)d_in[12];
  const float* gate_w   = (const float*)d_in[13];
  const float* up_w     = (const float*)d_in[14];
  const float* down_w   = (const float*)d_in[15];
  const float* sg_w     = (const float*)d_in[16];
  const float* su_w     = (const float*)d_in[17];
  const float* sd_w     = (const float*)d_in[18];
  float* out = (float*)d_out;

  float* f = (float*)d_ws;
  float* qa    = f; f += (size_t)S_LEN * QR_;
  float* kva   = f; f += (size_t)S_LEN * KVR_;
  float* qbuf  = f; f += (size_t)S_LEN * NH_ * DQK_;
  float* kvbuf = f; f += (size_t)S_LEN * NH_ * (DQK_ + DV_);
  float* x1    = f; f += (size_t)S_LEN * H_DIM;
  float* comb  = f; f += (size_t)S_LEN * NE_;
  float* g = qbuf;
  float* u = qbuf + (size_t)S_LEN * MMOE_;

  _Float16* hf = (_Float16*)f;
  _Float16* h1f   = hf; hf += (size_t)S_LEN * H_DIM;
  _Float16* qa16  = hf; hf += (size_t)S_LEN * QR_;
  _Float16* kva16 = hf; hf += (size_t)S_LEN * KVR_;
  _Float16* ctx16 = hf; hf += (size_t)S_LEN * NH_ * DV_;
  _Float16* act16 = hf; hf += (size_t)S_LEN * MMOE_;
  _Float16* qaw16  = hf; hf += (size_t)QR_ * H_DIM;
  _Float16* kvaw16 = hf; hf += (size_t)KVR_ * H_DIM;
  _Float16* qbw16  = hf; hf += (size_t)NH_ * DQK_ * QR_;
  _Float16* kvbw16 = hf; hf += (size_t)NH_ * (DQK_ + DV_) * KVR_;
  _Float16* ow16   = hf; hf += (size_t)H_DIM * NH_ * DV_;
  _Float16* sgw16  = hf; hf += (size_t)MMOE_ * H_DIM;
  _Float16* suw16  = hf; hf += (size_t)MMOE_ * H_DIM;
  _Float16* sdw16  = hf; hf += (size_t)H_DIM * MMOE_;
  _Float16* ge16   = hf; hf += (size_t)MMOE_ * H_DIM;
  _Float16* ue16   = hf; hf += (size_t)MMOE_ * H_DIM;
  _Float16* de16   = hf; hf += (size_t)H_DIM * MMOE_;

  dim3 blk(256);
  auto gemm = [&](const _Float16* A, const _Float16* B, float* C,
                  const float* res, int M, int N, int K, int mode) {
    dim3 grid(N / GBN, M / GBM);
    gemm_nt_f16_kernel<<<grid, blk, 0, stream>>>(A, B, C, res, M, N, K, mode);
  };
  auto cvt = [&](const float* in, _Float16* o16, size_t n) {
    cvt_f16_kernel<<<(unsigned)(n / 8 + 255) / 256, 256, 0, stream>>>(in, o16, (int)n);
  };

  cvt(q_a_w,  qaw16,  (size_t)QR_ * H_DIM);
  cvt(kv_a_w, kvaw16, (size_t)KVR_ * H_DIM);
  cvt(q_b_w,  qbw16,  (size_t)NH_ * DQK_ * QR_);
  cvt(kv_b_w, kvbw16, (size_t)NH_ * (DQK_ + DV_) * KVR_);
  cvt(o_w,    ow16,   (size_t)H_DIM * NH_ * DV_);
  cvt(sg_w,   sgw16,  (size_t)MMOE_ * H_DIM);
  cvt(su_w,   suw16,  (size_t)MMOE_ * H_DIM);
  cvt(sd_w,   sdw16,  (size_t)H_DIM * MMOE_);

  rmsnorm_f16_kernel<<<S_LEN, 256, 0, stream>>>(x, ln1_w, h1f, H_DIM);
  gemm(h1f, qaw16,  qa,  nullptr, S_LEN, QR_,  H_DIM, 0);
  gemm(h1f, kvaw16, kva, nullptr, S_LEN, KVR_, H_DIM, 0);
  rmsnorm_f16_kernel<<<S_LEN, 256, 0, stream>>>(qa,  q_a_ln,  qa16,  QR_);
  rmsnorm_f16_kernel<<<S_LEN, 256, 0, stream>>>(kva, kv_a_ln, kva16, KVR_);
  gemm(qa16,  qbw16,  qbuf,  nullptr, S_LEN, NH_ * DQK_,         QR_,  0);
  gemm(kva16, kvbw16, kvbuf, nullptr, S_LEN, NH_ * (DQK_ + DV_), KVR_, 0);
  rope_kernel<<<dim3(NH_, S_LEN), 64, 0, stream>>>(qbuf, kvbuf);
  flash_mla_kernel<<<dim3(NH_, S_LEN / 128), 256, 0, stream>>>(
      qbuf, kvbuf, ctx16, 1.0f / sqrtf((float)DQK_));
  gemm(ctx16, ow16, x1, x, S_LEN, H_DIM, NH_ * DV_, 1);

  rmsnorm_f16_kernel<<<S_LEN, 256, 0, stream>>>(x1, ln2_w, h1f, H_DIM);
  router_kernel<<<S_LEN / 128, 128, 0, stream>>>(h1f, router_w, router_b,
                                                 comb, S_LEN, H_DIM);
  gemm(h1f, sgw16, g, nullptr, S_LEN, MMOE_, H_DIM, 0);
  gemm(h1f, suw16, u, nullptr, S_LEN, MMOE_, H_DIM, 0);
  silu_mul_kernel<<<(S_LEN * MMOE_) / 256, 256, 0, stream>>>(
      g, u, nullptr, 0, act16, S_LEN * MMOE_, MMOE_);
  gemm(act16, sdw16, out, x1, S_LEN, H_DIM, MMOE_, 1);
  for (int e = 0; e < NE_; ++e) {
    cvt(gate_w + (size_t)e * MMOE_ * H_DIM, ge16, (size_t)MMOE_ * H_DIM);
    cvt(up_w   + (size_t)e * MMOE_ * H_DIM, ue16, (size_t)MMOE_ * H_DIM);
    cvt(down_w + (size_t)e * H_DIM * MMOE_, de16, (size_t)H_DIM * MMOE_);
    gemm(h1f, ge16, g, nullptr, S_LEN, MMOE_, H_DIM, 0);
    gemm(h1f, ue16, u, nullptr, S_LEN, MMOE_, H_DIM, 0);
    silu_mul_kernel<<<(S_LEN * MMOE_) / 256, 256, 0, stream>>>(
        g, u, comb, e, act16, S_LEN * MMOE_, MMOE_);
    gemm(act16, de16, out, nullptr, S_LEN, H_DIM, MMOE_, 2);
  }
}
